// NodeEmbDecoder_86801289052246
// MI455X (gfx1250) — matrix operand order repaired and checked by reading, not compiled, not run on hardware
//
#include <hip/hip_runtime.h>
#include <hip/hip_bf16.h>

typedef __bf16 bf16_t;
typedef __attribute__((ext_vector_type(16))) __bf16 v16bf;
typedef __attribute__((ext_vector_type(8)))  float  v8f;
typedef __attribute__((ext_vector_type(4)))  unsigned int u32x4;

#define HID   512
#define LH    1536      // 3*HID
#define BATCH 2048
#define TSTEPS 64
#define NDIM  256

union FragU { v16bf v; u32x4 u[2]; };

// A fragment: 16x32 bf16 from row-major [M,K] (lda elems). ISA 7.12.2 layout:
// lane L: m = m0 + (L&15), half = L>>4; VGPR0-3 hold K = half*8+0..7,
// VGPR4-7 hold K = 16+half*8+0..7  -> two contiguous 16B chunks.
__device__ __forceinline__ v16bf load_a_frag(const bf16_t* A, int lda, int m0, int k0, int lane) {
  int m    = m0 + (lane & 15);
  int half = lane >> 4;
  const bf16_t* p = A + (size_t)m * lda + k0 + half * 8;
  FragU r;
  r.u[0] = *(const u32x4*)p;
  r.u[1] = *(const u32x4*)(p + 16);
  return r.v;
}

// B fragment: 32x16 bf16 for D=A*B with B[k,n] = W[n,k], W row-major [N,K].
// lane L: n = n0 + (L&15), half = L>>4; VGPR0-3 hold K = half*8+0..7,
// VGPR4-7 hold K = 16+half*8+0..7 (same K order as the A fragment).
__device__ __forceinline__ v16bf load_b_frag(const bf16_t* W, int ldw, int n0, int k0, int lane) {
  int n    = n0 + (lane & 15);
  int half = lane >> 4;
  const bf16_t* p = W + (size_t)n * ldw + k0 + half * 8;
  FragU r;
  r.u[0] = *(const u32x4*)p;
  r.u[1] = *(const u32x4*)(p + 16);
  return r.v;
}

__device__ __forceinline__ float sigm_(float x) { return 1.f / (1.f + __expf(-x)); }

// ---------------------------------------------------------------------------
// Generic bf16 WMMA GEMM: out = act(A[M,K] * W[N,K]^T + bias)
// One wave per 32x32 output tile (2x2 of 16x16 WMMA tiles): 4 frag loads feed
// 4 WMMAs per K-step. K multiple of 32; M,N multiples of 32.
// ---------------------------------------------------------------------------
__global__ __launch_bounds__(256) void gemm_bf16_kernel(
    const bf16_t* __restrict__ A, int lda,
    const bf16_t* __restrict__ W,            // [N,K], ldw == K
    const float*  __restrict__ bias,
    float* outF, long long outF_rstride,
    bf16_t* outB, int outB_rstride,
    int K, int ntile_n2, int relu)           // ntile_n2 = N/32
{
  const int tid  = threadIdx.x;
  const int lane = tid & 31;
  const int wave = tid >> 5;
  const int tile = blockIdx.x * 8 + wave;
  const int mt = tile / ntile_n2;
  const int nt = tile - mt * ntile_n2;
  const int m0 = mt * 32, n0 = nt * 32;

  v8f acc[2][2];
#pragma unroll
  for (int mi = 0; mi < 2; ++mi)
#pragma unroll
    for (int ni = 0; ni < 2; ++ni)
      acc[mi][ni] = (v8f){0.f,0.f,0.f,0.f,0.f,0.f,0.f,0.f};

  for (int k0 = 0; k0 < K; k0 += 32) {
    v16bf a0 = load_a_frag(A, lda, m0,      k0, lane);
    v16bf a1 = load_a_frag(A, lda, m0 + 16, k0, lane);
    v16bf b0 = load_b_frag(W, K,  n0,      k0, lane);
    v16bf b1 = load_b_frag(W, K,  n0 + 16, k0, lane);
    acc[0][0] = __builtin_amdgcn_wmma_f32_16x16x32_bf16(false, a0, false, b0, (short)0, acc[0][0], false, false);
    acc[0][1] = __builtin_amdgcn_wmma_f32_16x16x32_bf16(false, a0, false, b1, (short)0, acc[0][1], false, false);
    acc[1][0] = __builtin_amdgcn_wmma_f32_16x16x32_bf16(false, a1, false, b0, (short)0, acc[1][0], false, false);
    acc[1][1] = __builtin_amdgcn_wmma_f32_16x16x32_bf16(false, a1, false, b1, (short)0, acc[1][1], false, false);
  }

  const int halfm = lane >> 4;
#pragma unroll
  for (int ni = 0; ni < 2; ++ni) {
    const int n  = n0 + ni * 16 + (lane & 15);
    const float bv = bias[n];
#pragma unroll
    for (int mi = 0; mi < 2; ++mi) {
#pragma unroll
      for (int r = 0; r < 8; ++r) {
        const int m = m0 + mi * 16 + r + 8 * halfm;
        float v = acc[mi][ni][r] + bv;
        if (relu) v = v > 0.f ? v : 0.f;
        if (outF) outF[(long long)m * outF_rstride + n] = v;
        if (outB) outB[(size_t)m * outB_rstride + n] = (bf16_t)v;
      }
    }
  }
}

// ---------------------------------------------------------------------------
// Fused LSTM step (one layer): gates = X*Wih^T + Hin*Whh^T + bih + bhh,
// then cell update. Each wave owns 32 rows x 16 hidden cols and accumulates
// ALL FOUR gates (cols j, 512+j, 1024+j, 1536+j): 8 WMMAs per K-step fed by
// 2 A-frag + 4 B-frag loads. Hin/Hout/C are column-slices of the
// [2048,1536] state buffers (ld = LH).
// ---------------------------------------------------------------------------
__global__ __launch_bounds__(256) void lstm_step_kernel(
    const bf16_t* __restrict__ X, int ldx,
    const bf16_t* __restrict__ Hin,          // ld = LH
    bf16_t* __restrict__ Hout,               // ld = LH
    const bf16_t* __restrict__ Wih,          // [4*HID, HID]
    const bf16_t* __restrict__ Whh,          // [4*HID, HID]
    const float* __restrict__ bih, const float* __restrict__ bhh,
    float* __restrict__ C)                   // ld = LH
{
  const int tid  = threadIdx.x;
  const int lane = tid & 31;
  const int wave = tid >> 5;
  const int tile = blockIdx.x * 8 + wave;    // 64 m-tiles(32) x 32 j-tiles(16)
  const int mt = tile >> 5;
  const int jt = tile & 31;
  const int m0 = mt * 32, j0 = jt * 16;

  v8f acc[2][4];
#pragma unroll
  for (int mi = 0; mi < 2; ++mi)
#pragma unroll
    for (int g = 0; g < 4; ++g)
      acc[mi][g] = (v8f){0.f,0.f,0.f,0.f,0.f,0.f,0.f,0.f};

  for (int k0 = 0; k0 < HID; k0 += 32) {
    v16bf a0 = load_a_frag(X, ldx, m0,      k0, lane);
    v16bf a1 = load_a_frag(X, ldx, m0 + 16, k0, lane);
#pragma unroll
    for (int g = 0; g < 4; ++g) {
      v16bf b = load_b_frag(Wih, HID, g * HID + j0, k0, lane);
      acc[0][g] = __builtin_amdgcn_wmma_f32_16x16x32_bf16(false, a0, false, b, (short)0, acc[0][g], false, false);
      acc[1][g] = __builtin_amdgcn_wmma_f32_16x16x32_bf16(false, a1, false, b, (short)0, acc[1][g], false, false);
    }
  }
  for (int k0 = 0; k0 < HID; k0 += 32) {
    v16bf a0 = load_a_frag(Hin, LH, m0,      k0, lane);
    v16bf a1 = load_a_frag(Hin, LH, m0 + 16, k0, lane);
#pragma unroll
    for (int g = 0; g < 4; ++g) {
      v16bf b = load_b_frag(Whh, HID, g * HID + j0, k0, lane);
      acc[0][g] = __builtin_amdgcn_wmma_f32_16x16x32_bf16(false, a0, false, b, (short)0, acc[0][g], false, false);
      acc[1][g] = __builtin_amdgcn_wmma_f32_16x16x32_bf16(false, a1, false, b, (short)0, acc[1][g], false, false);
    }
  }

  const int j     = j0 + (lane & 15);
  const int halfm = lane >> 4;
  float bg[4];
#pragma unroll
  for (int g = 0; g < 4; ++g) bg[g] = bih[g * HID + j] + bhh[g * HID + j];

#pragma unroll
  for (int mi = 0; mi < 2; ++mi) {
#pragma unroll
    for (int r = 0; r < 8; ++r) {
      const int m = m0 + mi * 16 + r + 8 * halfm;
      const float gi = acc[mi][0][r] + bg[0];
      const float gf = acc[mi][1][r] + bg[1];
      const float gg = acc[mi][2][r] + bg[2];
      const float go = acc[mi][3][r] + bg[3];
      const size_t idx = (size_t)m * LH + j;
      const float c_old = C[idx];
      const float c_new = sigm_(gf) * c_old + sigm_(gi) * tanhf(gg);
      const float h_new = sigm_(go) * tanhf(c_new);
      C[idx]    = c_new;
      Hout[idx] = (bf16_t)h_new;
    }
  }
}

// ---------------------------------------------------------------------------
__global__ void f2bf_kernel(const float* __restrict__ in, bf16_t* __restrict__ out, int n) {
  int i = blockIdx.x * 256 + threadIdx.x;
  if (i < n) out[i] = (bf16_t)in[i];
}
__global__ void zero_u32_kernel(unsigned int* p, int n) {
  int i = blockIdx.x * 256 + threadIdx.x;
  if (i < n) p[i] = 0u;
}

// ---------------------------------------------------------------------------
extern "C" void kernel_launch(void* const* d_in, const int* in_sizes, int n_in,
                              void* d_out, int out_size, void* d_ws, size_t ws_size,
                              hipStream_t stream) {
  (void)in_sizes; (void)n_in; (void)out_size; (void)ws_size;
  const float* emb   = (const float*)d_in[0];
  const float* Win1  = (const float*)d_in[1];
  const float* bin1  = (const float*)d_in[2];
  const float* Win2  = (const float*)d_in[3];
  const float* bin2  = (const float*)d_in[4];
  const float* Wih   = (const float*)d_in[5];
  const float* Whh   = (const float*)d_in[6];
  const float* bih   = (const float*)d_in[7];
  const float* bhh   = (const float*)d_in[8];
  const float* Wout1 = (const float*)d_in[9];
  const float* bout1 = (const float*)d_in[10];
  const float* Wout2 = (const float*)d_in[11];
  const float* bout2 = (const float*)d_in[12];
  float* out = (float*)d_out;

  char* base = (char*)d_ws;
  size_t off = 0;
  auto alloc = [&](size_t bytes) -> void* {
    off = (off + 255) & ~(size_t)255;
    void* p = base + off;
    off += bytes;
    return p;
  };

  bf16_t* emb_bf   = (bf16_t*)alloc((size_t)BATCH * HID * 2);
  bf16_t* Win1_bf  = (bf16_t*)alloc((size_t)HID * HID * 2);
  bf16_t* Win2_bf  = (bf16_t*)alloc((size_t)LH * HID * 2);
  bf16_t* Wih_bf   = (bf16_t*)alloc((size_t)3 * 4 * HID * HID * 2);
  bf16_t* Whh_bf   = (bf16_t*)alloc((size_t)3 * 4 * HID * HID * 2);
  bf16_t* Wout1_bf = (bf16_t*)alloc((size_t)HID * HID * 2);
  bf16_t* Wout2_bf = (bf16_t*)alloc((size_t)NDIM * HID * 2);
  bf16_t* H1       = (bf16_t*)alloc((size_t)BATCH * HID * 2);
  bf16_t* hbuf0    = (bf16_t*)alloc((size_t)BATCH * LH * 2);
  bf16_t* hbuf1    = (bf16_t*)alloc((size_t)BATCH * LH * 2);
  float*  cbuf     = (float*) alloc((size_t)BATCH * LH * 4);
  bf16_t* x0       = (bf16_t*)alloc((size_t)BATCH * HID * 2);
  bf16_t* T1       = (bf16_t*)alloc((size_t)BATCH * HID * 2);

  auto cvt = [&](const float* s, bf16_t* d, int n) {
    f2bf_kernel<<<(n + 255) / 256, 256, 0, stream>>>(s, d, n);
  };
  cvt(emb,   emb_bf,   BATCH * HID);
  cvt(Win1,  Win1_bf,  HID * HID);
  cvt(Win2,  Win2_bf,  LH * HID);
  cvt(Wih,   Wih_bf,   3 * 4 * HID * HID);
  cvt(Whh,   Whh_bf,   3 * 4 * HID * HID);
  cvt(Wout1, Wout1_bf, HID * HID);
  cvt(Wout2, Wout2_bf, NDIM * HID);

  {
    int nc = BATCH * LH;                      // f32 words of c state
    zero_u32_kernel<<<(nc + 255) / 256, 256, 0, stream>>>((unsigned int*)cbuf, nc);
    int nx = BATCH * HID / 2;                 // u32 words of bf16 zeros
    zero_u32_kernel<<<(nx + 255) / 256, 256, 0, stream>>>((unsigned int*)x0, nx);
  }

  // FNN_in: H1 = relu(emb @ Win1^T + bin1) ; hx = H1 @ Win2^T + bin2 -> hbuf0
  // tiles32: GEMM1 (2048/32)*(512/32)=1024 waves -> 128 blocks
  gemm_bf16_kernel<<<128, 256, 0, stream>>>(emb_bf, HID, Win1_bf, bin1,
                                            nullptr, 0, H1, HID, HID, 16, 1);
  // GEMM2: (2048/32)*(1536/32)=3072 waves -> 384 blocks
  gemm_bf16_kernel<<<384, 256, 0, stream>>>(H1, HID, Win2_bf, bin2,
                                            nullptr, 0, hbuf0, LH, HID, 48, 0);

  bf16_t* hsrc = hbuf0;
  bf16_t* hdst = hbuf1;
  for (int t = 0; t < TSTEPS; ++t) {
    for (int l = 0; l < 3; ++l) {
      const bf16_t* X;
      int ldx;
      if (l == 0) {
        if (t == 0) { X = x0;               ldx = HID; }
        else        { X = hsrc + 2 * HID;   ldx = LH;  }   // prev-step layer-2 h
      } else        { X = hdst + (l - 1) * HID; ldx = LH; } // this-step layer l-1 h
      // 64 m-tiles(32 rows) x 32 j-tiles(16 cols) = 2048 waves -> 256 blocks
      lstm_step_kernel<<<256, 256, 0, stream>>>(
          X, ldx, hsrc + l * HID, hdst + l * HID,
          Wih_bf + (size_t)l * 4 * HID * HID,
          Whh_bf + (size_t)l * 4 * HID * HID,
          bih + l * 4 * HID, bhh + l * 4 * HID,
          cbuf + l * HID);
    }
    // FNN_out for this timestep's layer-2 output -> d_out rows b*64 + t
    gemm_bf16_kernel<<<128, 256, 0, stream>>>(hdst + 2 * HID, LH, Wout1_bf, bout1,
                                              nullptr, 0, T1, HID, HID, 16, 1);
    // (2048/32)*(256/32)=512 waves -> 64 blocks
    gemm_bf16_kernel<<<64, 256, 0, stream>>>(T1, HID, Wout2_bf, bout2,
                                             out + (size_t)t * NDIM,
                                             (long long)TSTEPS * NDIM,
                                             nullptr, 0, HID, 8, 0);
    bf16_t* tmp = hsrc; hsrc = hdst; hdst = tmp;  // ping-pong h state
  }
}
